// HATBlock_59889023975527
// MI455X (gfx1250) — matrix operand order repaired and checked by reading, not compiled, not run on hardware
//
#include <hip/hip_runtime.h>
#include <hip/hip_bf16.h>

typedef __attribute__((ext_vector_type(16))) _Float16 v16h;
typedef __attribute__((ext_vector_type(8)))  _Float16 v8h;
typedef __attribute__((ext_vector_type(8)))  float    v8f;

#define WAVES 8
#define TOKLD 65  // padded f32 token stride (bank-conflict-free transposed writes)

namespace {

__device__ __forceinline__ v8f wmma16(v16h a, v16h b, v8f c) {
  // D = A(16x32 f16) x B(32x16 f16) + C(16x16 f32)
  return __builtin_amdgcn_wmma_f32_16x16x32_f16(false, a, false, b, (short)0, c,
                                                false, false);
}

// A fragment (16x32 f16) from row-major LDS, leading dim ld (multiple of 16).
// Two contiguous 16B LDS loads per lane (K groups {kb..kb+7},{kb+16..kb+23}).
__device__ __forceinline__ v16h ldsA(const _Float16* base, int ld, int row0,
                                     int col0, int lane) {
  const int M  = lane & 15;
  const int kb = (lane >> 4) << 3;
  const _Float16* p = base + (row0 + M) * ld + col0 + kb;
  const v8h lo = *(const v8h*)p;
  const v8h hi = *(const v8h*)(p + 16);
  v16h a;
#pragma unroll
  for (int v = 0; v < 8; ++v) {
    a[v]     = lo[v];
    a[v + 8] = hi[v];
  }
  return a;
}

// B fragment (32x16) from a K-major f16 LDS matrix: B[k][n] = baseT[(k0+k)*N + n0+n].
// Element-strided LDS reads per lane (n = lane%16, K groups {kb..kb+7},{kb+16..kb+23}).
__device__ __forceinline__ v16h ldsBrow(const _Float16* baseT, int N, int k0,
                                        int n0, int lane) {
  const _Float16* p = baseT + (k0 + ((lane >> 4) << 3)) * N + n0 + (lane & 15);
  v16h b;
#pragma unroll
  for (int v = 0; v < 8; ++v) {
    b[v]     = p[v * N];
    b[v + 8] = p[(v + 16) * N];
  }
  return b;
}

// B fragment from a transposed LDS view (two contiguous 16B loads per lane):
// B[k][n] = base[(n0+n)*ld + c0+k].
__device__ __forceinline__ v16h ldsBT(const _Float16* base, int ld, int n0,
                                      int c0, int lane) {
  const _Float16* p = base + (n0 + (lane & 15)) * ld + c0 + ((lane >> 4) << 3);
  const v8h lo = *(const v8h*)p;
  const v8h hi = *(const v8h*)(p + 16);
  v16h b;
#pragma unroll
  for (int v = 0; v < 8; ++v) {
    b[v]     = lo[v];
    b[v + 8] = hi[v];
  }
  return b;
}

// Stage W (N x K f32, row-major, global/L2) -> WT (K x N f16, LDS).
// Global reads coalesced along k.
__device__ __forceinline__ void stageWT(const float* __restrict__ W,
                                        _Float16* WT, int N, int K, int tid) {
  for (int i = tid; i < N * K; i += 256) {
    const int n = i / K, k = i - n * K;
    WT[k * N + n] = (_Float16)W[i];
  }
}

}  // namespace

__global__ __launch_bounds__(256) void hat_block_kernel(
    const float* __restrict__ x,
    const float* __restrict__ n1g, const float* __restrict__ n1b,
    const float* __restrict__ in_w, const float* __restrict__ in_b,
    const float* __restrict__ out_w, const float* __restrict__ out_b,
    const float* __restrict__ proj_w, const float* __restrict__ proj_b,
    const float* __restrict__ n2g, const float* __restrict__ n2b,
    const float* __restrict__ w1, const float* __restrict__ b1,
    const float* __restrict__ w2, const float* __restrict__ b2,
    float* __restrict__ out) {
  __shared__ float sTok[64 * TOKLD];                  // tokens f32 (residual)
  __shared__ __align__(64) _Float16 arena[24576];     // 48 KB, stage-reused
  // f16 transposed weights, staged once per block (L2 -> LDS), ~72 KB
  __shared__ __align__(64) _Float16 sInWT[64 * 192];
  __shared__ __align__(64) _Float16 sOutWT[64 * 64];
  __shared__ __align__(64) _Float16 sProjWT[64 * 64];
  __shared__ __align__(64) _Float16 sW1T[64 * 128];
  __shared__ __align__(64) _Float16 sW2T[128 * 64];

  _Float16* sTn   = arena;            // 64x64   (LN1 out)
  _Float16* sQKV  = arena + 4096;     // 64x192
  _Float16* sAttn = arena + 16384;    // 2x64x64 (softmaxed attn, f16)
  _Float16* sO1   = arena;            // 64x64   (reuse: sTn dead)
  _Float16* sO2   = arena + 4096;     // 64x64   (reuse: sQKV dead)
  _Float16* sTn2  = arena;            // 64x64   (reuse: sO1 dead)
  _Float16* sH    = arena + 8192;     // 64x128  (reuse: sQKV tail dead)

  const int tid  = threadIdx.x;
  const int lane = tid & 31;
  const int wave = tid >> 5;

  const int wid = blockIdx.x;  // window id, 0..16383
  const int bb  = wid >> 12;
  const int wh  = (wid >> 6) & 63;
  const int ww  = wid & 63;

  // ---- stage weights as f16, K-major, into LDS (global reads coalesced) ----
  stageWT(in_w,   sInWT,  192, 64,  tid);
  stageWT(out_w,  sOutWT, 64,  64,  tid);
  stageWT(proj_w, sProjWT,64,  64,  tid);
  stageWT(w1,     sW1T,   128, 64,  tid);
  stageWT(w2,     sW2T,   64,  128, tid);

  // ---- gather window: (B,C,H,W) -> tokens; lanes walk wp fastest (coalesced)
  for (int i = tid; i < 4096; i += 256) {
    const int c = i >> 6, t = i & 63;
    const int hh = (wh << 3) + (t >> 3);
    const int wp = (ww << 3) + (t & 7);
    sTok[t * TOKLD + c] = x[(((size_t)bb * 64 + c) * 512 + hh) * 512 + wp];
  }
  __syncthreads();

  // ---- LayerNorm 1 -> sTn (f16) ----
  if (tid < 64) {
    const float* row = sTok + tid * TOKLD;
    float m = 0.f;
    for (int c = 0; c < 64; ++c) m += row[c];
    m *= (1.f / 64.f);
    float v = 0.f;
    for (int c = 0; c < 64; ++c) { const float d = row[c] - m; v += d * d; }
    const float inv = rsqrtf(v * (1.f / 64.f) + 1e-5f);
    for (int c = 0; c < 64; ++c)
      sTn[tid * 64 + c] = (_Float16)((row[c] - m) * inv * n1g[c] + n1b[c]);
  }
  __syncthreads();

  // ---- QKV = tn @ in_w^T + in_b -> sQKV (64 x 192) : 48 tiles / 8 waves ----
  for (int t = wave; t < 48; t += WAVES) {
    const int rt = t & 3, ct = t >> 2;
    v8f c = {};
#pragma unroll
    for (int kk = 0; kk < 64; kk += 32)
      c = wmma16(ldsA(sTn, 64, rt * 16, kk, lane),
                 ldsBrow(sInWT, 192, kk, ct * 16, lane), c);
    const int N = lane & 15, Mb = (lane >> 4) << 3;
    const float bias = in_b[ct * 16 + N];
    _Float16* p = sQKV + (rt * 16 + Mb) * 192 + ct * 16 + N;
#pragma unroll
    for (int r = 0; r < 8; ++r) p[r * 192] = (_Float16)(c[r] + bias);
  }
  __syncthreads();

  // ---- scores strip (16x64) per (head, row-tile) wave; softmax in regs ----
  {
    const int h = wave >> 2, rt = wave & 3;
    v8f acc[4];
    const v16h aq = ldsA(sQKV, 192, rt * 16, h * 32, lane);  // q, K=32
#pragma unroll
    for (int ct = 0; ct < 4; ++ct) {
      v8f z = {};
      acc[ct] = wmma16(aq, ldsBT(sQKV, 192, ct * 16, 64 + h * 32, lane), z);
    }
    const float scale = 0.17677669529663687f;  // 1/sqrt(32)
    const int N = lane & 15, Mb = (lane >> 4) << 3;
#pragma unroll
    for (int r = 0; r < 8; ++r) {
      float m = -3.0e38f;
#pragma unroll
      for (int ct = 0; ct < 4; ++ct) {
        acc[ct][r] *= scale;
        m = fmaxf(m, acc[ct][r]);
      }
#pragma unroll
      for (int off = 1; off < 16; off <<= 1) m = fmaxf(m, __shfl_xor(m, off, 32));
      float s = 0.f;
#pragma unroll
      for (int ct = 0; ct < 4; ++ct) {
        const float e = __expf(acc[ct][r] - m);
        acc[ct][r] = e;
        s += e;
      }
#pragma unroll
      for (int off = 1; off < 16; off <<= 1) s += __shfl_xor(s, off, 32);
      const float inv = 1.f / s;
      _Float16* p = sAttn + (h * 64 + rt * 16 + Mb + r) * 64 + N;
#pragma unroll
      for (int ct = 0; ct < 4; ++ct) p[ct * 16] = (_Float16)(acc[ct][r] * inv);
    }
  }
  __syncthreads();

  // ---- o1 = attn @ v (per head) -> sO1 (64x64) ----
  {
    const int h = wave >> 2, rt = wave & 3;
#pragma unroll
    for (int ct = 0; ct < 2; ++ct) {
      v8f c = {};
#pragma unroll
      for (int kk = 0; kk < 64; kk += 32)
        c = wmma16(ldsA(sAttn + h * 4096, 64, rt * 16, kk, lane),
                   ldsBrow(sQKV, 192, kk, 128 + h * 32 + ct * 16, lane), c);
      const int N = lane & 15, Mb = (lane >> 4) << 3;
      _Float16* p = sO1 + (rt * 16 + Mb) * 64 + h * 32 + ct * 16 + N;
#pragma unroll
      for (int r = 0; r < 8; ++r) p[r * 64] = (_Float16)c[r];
    }
  }
  __syncthreads();

  // ---- o2 = o1 @ out_w^T + out_b -> sO2 ----
  for (int t = wave; t < 16; t += WAVES) {
    const int rt = t & 3, ct = t >> 2;
    v8f c = {};
#pragma unroll
    for (int kk = 0; kk < 64; kk += 32)
      c = wmma16(ldsA(sO1, 64, rt * 16, kk, lane),
                 ldsBrow(sOutWT, 64, kk, ct * 16, lane), c);
    const int N = lane & 15, Mb = (lane >> 4) << 3;
    const float bias = out_b[ct * 16 + N];
    _Float16* p = sO2 + (rt * 16 + Mb) * 64 + ct * 16 + N;
#pragma unroll
    for (int r = 0; r < 8; ++r) p[r * 64] = (_Float16)(c[r] + bias);
  }
  __syncthreads();

  // ---- tokens += o2 @ proj_w^T + proj_b ----
  for (int t = wave; t < 16; t += WAVES) {
    const int rt = t & 3, ct = t >> 2;
    v8f c = {};
#pragma unroll
    for (int kk = 0; kk < 64; kk += 32)
      c = wmma16(ldsA(sO2, 64, rt * 16, kk, lane),
                 ldsBrow(sProjWT, 64, kk, ct * 16, lane), c);
    const int N = lane & 15, Mb = (lane >> 4) << 3;
    const float bias = proj_b[ct * 16 + N];
    float* p = sTok + (rt * 16 + Mb) * TOKLD + ct * 16 + N;
#pragma unroll
    for (int r = 0; r < 8; ++r) p[r * TOKLD] += c[r] + bias;
  }
  __syncthreads();

  // ---- LayerNorm 2 -> sTn2 (f16) ----
  if (tid < 64) {
    const float* row = sTok + tid * TOKLD;
    float m = 0.f;
    for (int c = 0; c < 64; ++c) m += row[c];
    m *= (1.f / 64.f);
    float v = 0.f;
    for (int c = 0; c < 64; ++c) { const float d = row[c] - m; v += d * d; }
    const float inv = rsqrtf(v * (1.f / 64.f) + 1e-5f);
    for (int c = 0; c < 64; ++c)
      sTn2[tid * 64 + c] = (_Float16)((row[c] - m) * inv * n2g[c] + n2b[c]);
  }
  __syncthreads();

  // ---- hidden = gelu(tn2 @ w1^T + b1) -> sH (64 x 128) ----
  for (int t = wave; t < 32; t += WAVES) {
    const int rt = t & 3, ct = t >> 2;  // ct 0..7
    v8f c = {};
#pragma unroll
    for (int kk = 0; kk < 64; kk += 32)
      c = wmma16(ldsA(sTn2, 64, rt * 16, kk, lane),
                 ldsBrow(sW1T, 128, kk, ct * 16, lane), c);
    const int N = lane & 15, Mb = (lane >> 4) << 3;
    const float bias = b1[ct * 16 + N];
    _Float16* p = sH + (rt * 16 + Mb) * 128 + ct * 16 + N;
#pragma unroll
    for (int r = 0; r < 8; ++r) {
      const float vv = c[r] + bias;
      const float g = 0.5f * vv * (1.f + erff(vv * 0.70710678118654752f));
      p[r * 128] = (_Float16)g;
    }
  }
  __syncthreads();

  // ---- tokens += H @ w2^T + b2 ----
  for (int t = wave; t < 16; t += WAVES) {
    const int rt = t & 3, ct = t >> 2;
    v8f c = {};
#pragma unroll
    for (int kk = 0; kk < 128; kk += 32)
      c = wmma16(ldsA(sH, 128, rt * 16, kk, lane),
                 ldsBrow(sW2T, 64, kk, ct * 16, lane), c);
    const int N = lane & 15, Mb = (lane >> 4) << 3;
    const float bias = b2[ct * 16 + N];
    float* p = sTok + (rt * 16 + Mb) * TOKLD + ct * 16 + N;
#pragma unroll
    for (int r = 0; r < 8; ++r) p[r * TOKLD] += c[r] + bias;
  }
  __syncthreads();

  // ---- window reverse: tokens -> (B,C,H,W), lanes walk wp fastest ----
  for (int i = tid; i < 4096; i += 256) {
    const int c = i >> 6, t = i & 63;
    const int hh = (wh << 3) + (t >> 3);
    const int wp = (ww << 3) + (t & 7);
    out[(((size_t)bb * 64 + c) * 512 + hh) * 512 + wp] = sTok[t * TOKLD + c];
  }
}

extern "C" void kernel_launch(void* const* d_in, const int* in_sizes, int n_in,
                              void* d_out, int out_size, void* d_ws, size_t ws_size,
                              hipStream_t stream) {
  (void)in_sizes; (void)n_in; (void)out_size; (void)d_ws; (void)ws_size;
  hat_block_kernel<<<dim3(16384), dim3(256), 0, stream>>>(
      (const float*)d_in[0],  (const float*)d_in[1],  (const float*)d_in[2],
      (const float*)d_in[3],  (const float*)d_in[4],  (const float*)d_in[5],
      (const float*)d_in[6],  (const float*)d_in[7],  (const float*)d_in[8],
      (const float*)d_in[9],  (const float*)d_in[10], (const float*)d_in[11],
      (const float*)d_in[12], (const float*)d_in[13], (const float*)d_in[14],
      (float*)d_out);
}
